// Attention_61168924230279
// MI455X (gfx1250) — matrix operand order repaired and checked by reading, not compiled, not run on hardware
//
#include <hip/hip_runtime.h>
#include <hip/hip_bf16.h>

// ---------------------------------------------------------------------------
// MHA forward for MI455X (gfx1250, wave32, WMMA).
// All GEMMs (QKV proj, QK^T, P*V, out proj) run on v_wmma_f32_16x16x32_bf16.
// Attention uses flash-style online softmax; P tile is re-laid-out C->A via a
// small per-wave LDS staging buffer.
// GEMM wave tile is 32(M)x64(N): every B-frag is reused by two A-frags in
// registers (8 WMMAs per 12 b128 loads per 32-k step).
// ---------------------------------------------------------------------------

#define B_   2
#define S_   2048
#define D_   2048
#define H_   16
#define HD_  128
#define M_   (B_ * S_)          // 4096 rows for all projection GEMMs

typedef __attribute__((ext_vector_type(16))) __bf16 v16bf;
typedef __attribute__((ext_vector_type(8)))  __bf16 v8bf;
typedef __attribute__((ext_vector_type(8)))  float  v8f;

static __device__ __forceinline__ v8f wmma_bf16(v16bf a, v16bf b, v8f c) {
    // D = A(16x32 bf16) * B(32x16 bf16) + C(16x16 f32)
    return __builtin_amdgcn_wmma_f32_16x16x32_bf16(
        /*neg_a=*/false, a, /*neg_b=*/false, b,
        /*c_mod=*/(short)0, c, /*reuse_a=*/false, /*reuse_b=*/false);
}

static __device__ __forceinline__ v16bf concat8(v8bf lo, v8bf hi) {
    return __builtin_shufflevector(lo, hi, 0,1,2,3,4,5,6,7,8,9,10,11,12,13,14,15);
}

// ---------------------------------------------------------------------------
// fp32 -> bf16 cast
// ---------------------------------------------------------------------------
__global__ void cast_bf16_kernel(const float* __restrict__ in,
                                 __bf16* __restrict__ out, int n) {
    int i = blockIdx.x * blockDim.x + threadIdx.x;
    if (i < n) out[i] = (__bf16)in[i];
}

// ---------------------------------------------------------------------------
// C[M,N] (f32) = A[M,K](bf16) * W[N,K](bf16)^T + bias[N]
// M=4096, N=K=2048. Block = 256 thr = 8 waves. Wave tile = 32(M) x 64(N).
// Block tile = 256(M) x 64(N).  Grid = (N/64, M/256) = (32, 16).
// ---------------------------------------------------------------------------
__global__ __launch_bounds__(256)
void gemm_bias_kernel(const __bf16* __restrict__ A,
                      const __bf16* __restrict__ W,
                      const float* __restrict__ bias,
                      float* __restrict__ C) {
    const int lane  = threadIdx.x & 31;
    const int wave  = threadIdx.x >> 5;
    const int lm    = lane & 15;
    const int kA    = (lane < 16) ? 0 : 16;   // A-frag K sub-offset
    const int kB    = (lane < 16) ? 0 : 16;   // B-frag K sub-offset
    const int half8 = (lane < 16) ? 0 : 8;    // C-layout row offset

    const int m0 = blockIdx.y * 256 + wave * 32;
    const int n0 = blockIdx.x * 64;

    v8f acc[2][4];
#pragma unroll
    for (int u = 0; u < 2; ++u)
#pragma unroll
        for (int t = 0; t < 4; ++t)
#pragma unroll
            for (int r = 0; r < 8; ++r) acc[u][t][r] = 0.0f;

    const __bf16* arow0 = A + (size_t)(m0 + lm) * D_;
    const __bf16* arow1 = A + (size_t)(m0 + 16 + lm) * D_;

    for (int k0 = 0; k0 < D_; k0 += 32) {
        v16bf af0 = concat8(*(const v8bf*)(arow0 + k0 + kA),
                            *(const v8bf*)(arow0 + k0 + kA + 8));
        v16bf af1 = concat8(*(const v8bf*)(arow1 + k0 + kA),
                            *(const v8bf*)(arow1 + k0 + kA + 8));
#pragma unroll
        for (int t = 0; t < 4; ++t) {
            const __bf16* wrow = W + (size_t)(n0 + 16 * t + lm) * D_ + k0 + kB;
            v16bf bf = *(const v16bf*)wrow;   // 16 contiguous K values
            acc[0][t] = wmma_bf16(af0, bf, acc[0][t]);
            acc[1][t] = wmma_bf16(af1, bf, acc[1][t]);
        }
    }

#pragma unroll
    for (int t = 0; t < 4; ++t) {
        const int ncol = n0 + 16 * t + lm;
        const float bv = bias[ncol];
#pragma unroll
        for (int r = 0; r < 8; ++r) {
            C[(size_t)(m0 + r + half8) * D_ + ncol]      = acc[0][t][r] + bv;
            C[(size_t)(m0 + 16 + r + half8) * D_ + ncol] = acc[1][t][r] + bv;
        }
    }
}

// ---------------------------------------------------------------------------
// RoPE: in f32 (B,S,D) -> out bf16 (B,S,D).  One thread per (b,s,h,i<64) pair.
// ---------------------------------------------------------------------------
__global__ void rope_bf16_kernel(const float* __restrict__ in,
                                 __bf16* __restrict__ out) {
    int idx = blockIdx.x * blockDim.x + threadIdx.x;   // B*S*H*64 threads
    const int i = idx & 63;
    const int h = (idx >> 6) & (H_ - 1);
    const int s = (idx >> 10) & (S_ - 1);
    const int b = idx >> 21;
    const size_t base = ((size_t)(b * S_ + s)) * D_ + h * HD_;
    const float t1 = in[base + i];
    const float t2 = in[base + i + 64];
    // inv_freq = theta^(-2i/hd) = 10000^(-i/64)
    const float inv = __powf(10000.0f, -(float)i * (1.0f / 64.0f));
    const float ang = (float)s * inv;
    float sn, cs;
    __sincosf(ang, &sn, &cs);
    out[base + i]      = (__bf16)(t1 * cs - t2 * sn);
    out[base + i + 64] = (__bf16)(t2 * cs + t1 * sn);
}

// ---------------------------------------------------------------------------
// V transpose: f32 (B,S,H,hd) -> bf16 Vt (B,H,hd,S)  (K-major for P*V B-frags)
// ---------------------------------------------------------------------------
__global__ void vtrans_kernel(const float* __restrict__ v,
                              __bf16* __restrict__ vt) {
    int idx = blockIdx.x * blockDim.x + threadIdx.x;   // B*S*D threads
    const int d = idx & (HD_ - 1);
    const int h = (idx >> 7) & (H_ - 1);
    const int s = (idx >> 11) & (S_ - 1);
    const int b = idx >> 22;
    const float val = v[((size_t)(b * S_ + s)) * D_ + h * HD_ + d];
    vt[((size_t)((b * H_ + h) * HD_ + d)) * S_ + s] = (__bf16)val;
}

// ---------------------------------------------------------------------------
// Flash attention.  Grid = (S/128, B*H).  Block = 256 thr = 8 waves.
// Each wave owns a 16-row Q tile, streams all 2048 keys in steps of 32,
// online softmax in VGPRs, O accumulated in 8 x (16x16) f32 WMMA tiles.
// ---------------------------------------------------------------------------
__global__ __launch_bounds__(256)
void attention_kernel(const __bf16* __restrict__ Q,
                      const __bf16* __restrict__ K,
                      const __bf16* __restrict__ Vt,
                      __bf16* __restrict__ O) {
    __shared__ __align__(32) __bf16 pls[8][16 * 32];   // per-wave P staging

    const int lane  = threadIdx.x & 31;
    const int wave  = threadIdx.x >> 5;
    const int lm    = lane & 15;
    const int kA    = (lane < 16) ? 0 : 16;
    const int kB    = (lane < 16) ? 0 : 16;
    const int half8 = (lane < 16) ? 0 : 8;

    const int bh = blockIdx.y;
    const int b  = bh >> 4;
    const int h  = bh & 15;
    const int q0 = blockIdx.x * 128 + wave * 16;

    // --- load Q fragments (16 rows x 128 head dims = 4 A-frags), kept resident
    v16bf qf[4];
    const __bf16* qrow = Q + ((size_t)(b * S_ + q0 + lm)) * D_ + h * HD_;
#pragma unroll
    for (int c = 0; c < 4; ++c) {
        v8bf lo = *(const v8bf*)(qrow + 32 * c + kA);
        v8bf hi = *(const v8bf*)(qrow + 32 * c + kA + 8);
        qf[c] = concat8(lo, hi);
    }

    v8f o[8];
    float mrow[8], lrow[8];
#pragma unroll
    for (int g = 0; g < 8; ++g)
#pragma unroll
        for (int r = 0; r < 8; ++r) o[g][r] = 0.0f;
#pragma unroll
    for (int r = 0; r < 8; ++r) { mrow[r] = -1.0e30f; lrow[r] = 0.0f; }

    const float scale = 0.08838834764831845f;   // 1/sqrt(128)

    for (int kk = 0; kk < S_; kk += 32) {
        // ---- scores: two 16x16 tiles (keys kk..kk+15, kk+16..kk+31)
        v8f s[2];
#pragma unroll
        for (int t = 0; t < 2; ++t) {
#pragma unroll
            for (int r = 0; r < 8; ++r) s[t][r] = 0.0f;
            const __bf16* krow =
                K + ((size_t)(b * S_ + kk + 16 * t + lm)) * D_ + h * HD_ + kB;
#pragma unroll
            for (int c = 0; c < 4; ++c) {
                v16bf kf = *(const v16bf*)(krow + 32 * c);
                s[t] = wmma_bf16(qf[c], kf, s[t]);
            }
        }
#pragma unroll
        for (int t = 0; t < 2; ++t)
#pragma unroll
            for (int r = 0; r < 8; ++r) s[t][r] *= scale;

        // ---- online softmax: row max / rescale / exp / row sum
        float alpha[8];
#pragma unroll
        for (int r = 0; r < 8; ++r) {
            float v = fmaxf(s[0][r], s[1][r]);
#pragma unroll
            for (int off = 8; off > 0; off >>= 1)
                v = fmaxf(v, __shfl_xor(v, off, 16));
            const float mnew = fmaxf(mrow[r], v);
            alpha[r] = __expf(mrow[r] - mnew);
            mrow[r]  = mnew;
        }
#pragma unroll
        for (int t = 0; t < 2; ++t)
#pragma unroll
            for (int r = 0; r < 8; ++r) s[t][r] = __expf(s[t][r] - mrow[r]);
#pragma unroll
        for (int r = 0; r < 8; ++r) {
            float rs = s[0][r] + s[1][r];
#pragma unroll
            for (int off = 8; off > 0; off >>= 1)
                rs += __shfl_xor(rs, off, 16);
            lrow[r] = lrow[r] * alpha[r] + rs;
        }

        // ---- stage P (C-layout) into LDS as row-major 16x32 bf16
        __syncthreads();   // WAR vs previous iteration's P-frag reads
#pragma unroll
        for (int t = 0; t < 2; ++t)
#pragma unroll
            for (int r = 0; r < 8; ++r)
                pls[wave][(r + half8) * 32 + 16 * t + lm] = (__bf16)s[t][r];
        __syncthreads();   // RAW: writes visible before A-layout reads

        // ---- rescale O by alpha
#pragma unroll
        for (int g = 0; g < 8; ++g)
#pragma unroll
            for (int r = 0; r < 8; ++r) o[g][r] *= alpha[r];

        // ---- P-frag (A layout) from LDS
        v8bf plo = *(const v8bf*)&pls[wave][lm * 32 + kA];
        v8bf phi = *(const v8bf*)&pls[wave][lm * 32 + kA + 8];
        v16bf pf = concat8(plo, phi);

        // ---- O += P(16x32) * V(32x16) over 8 head-dim column groups
        const __bf16* vbase =
            Vt + ((size_t)(bh * HD_ + lm)) * S_ + kk + kB;
#pragma unroll
        for (int g = 0; g < 8; ++g) {
            v16bf vf = *(const v16bf*)(vbase + (size_t)(16 * g) * S_);
            o[g] = wmma_bf16(pf, vf, o[g]);
        }
    }

    // ---- epilogue: O /= l, write bf16 (B,S,D)
    const size_t obase = ((size_t)(b * S_ + q0)) * D_ + h * HD_;
#pragma unroll
    for (int g = 0; g < 8; ++g)
#pragma unroll
        for (int r = 0; r < 8; ++r)
            O[obase + (size_t)(r + half8) * D_ + 16 * g + lm] =
                (__bf16)(o[g][r] / lrow[r]);
}

// ---------------------------------------------------------------------------
// Host-side orchestration
// ---------------------------------------------------------------------------
extern "C" void kernel_launch(void* const* d_in, const int* in_sizes, int n_in,
                              void* d_out, int out_size, void* d_ws, size_t ws_size,
                              hipStream_t stream) {
    const float* x  = (const float*)d_in[0];
    const float* wq = (const float*)d_in[1];
    const float* bq = (const float*)d_in[2];
    const float* wk = (const float*)d_in[3];
    const float* bk = (const float*)d_in[4];
    const float* wv = (const float*)d_in[5];
    const float* bv = (const float*)d_in[6];
    const float* wo = (const float*)d_in[7];
    const float* bo = (const float*)d_in[8];

    char* ws = (char*)d_ws;
    const size_t NX  = (size_t)M_ * D_;        // 8,388,608 elems
    const size_t NW  = (size_t)D_ * D_;        // 4,194,304 elems

    __bf16* XB  = (__bf16*)(ws);                                   // 16 MB
    __bf16* WQB = (__bf16*)(ws + NX * 2);                          //  8 MB
    __bf16* WKB = WQB + NW;
    __bf16* WVB = WKB + NW;
    __bf16* WOB = WVB + NW;
    float*  TF  = (float*) (ws + NX * 2 + 4 * NW * 2);             // 32 MB f32 temp
    __bf16* QB  = (__bf16*)((char*)TF + NX * 4);
    __bf16* KB  = QB + NX;
    __bf16* VTB = KB + NX;
    __bf16* OB  = VTB + NX;

    const dim3 blk(256);
    const dim3 gGemm(D_ / 64, M_ / 256);       // (32, 16)
    const dim3 gAttn(S_ / 128, B_ * H_);       // (16, 32)

    // 1. cast inputs to bf16
    cast_bf16_kernel<<<NX / 256, blk, 0, stream>>>(x,  XB,  (int)NX);
    cast_bf16_kernel<<<NW / 256, blk, 0, stream>>>(wq, WQB, (int)NW);
    cast_bf16_kernel<<<NW / 256, blk, 0, stream>>>(wk, WKB, (int)NW);
    cast_bf16_kernel<<<NW / 256, blk, 0, stream>>>(wv, WVB, (int)NW);
    cast_bf16_kernel<<<NW / 256, blk, 0, stream>>>(wo, WOB, (int)NW);

    // 2. Q = x*Wq^T + bq ; RoPE -> bf16
    gemm_bias_kernel<<<gGemm, blk, 0, stream>>>(XB, WQB, bq, TF);
    rope_bf16_kernel<<<(B_ * S_ * H_ * 64) / 256, blk, 0, stream>>>(TF, QB);

    // 3. K = x*Wk^T + bk ; RoPE -> bf16   (TF reused: stream-ordered)
    gemm_bias_kernel<<<gGemm, blk, 0, stream>>>(XB, WKB, bk, TF);
    rope_bf16_kernel<<<(B_ * S_ * H_ * 64) / 256, blk, 0, stream>>>(TF, KB);

    // 4. V = x*Wv^T + bv ; transpose to (B,H,hd,S) bf16
    gemm_bias_kernel<<<gGemm, blk, 0, stream>>>(XB, WVB, bv, TF);
    vtrans_kernel<<<NX / 256, blk, 0, stream>>>(TF, VTB);

    // 5. flash attention -> OB (bf16, B,S,D)
    attention_kernel<<<gAttn, blk, 0, stream>>>(QB, KB, VTB, OB);

    // 6. out = O*Wo^T + bo  (fp32 -> d_out)
    gemm_bias_kernel<<<gGemm, blk, 0, stream>>>(OB, WOB, bo, (float*)d_out);
}
